// CDDDDecoder_52055003627546
// MI455X (gfx1250) — matrix operand order repaired and checked by reading, not compiled, not run on hardware
//
#include <hip/hip_runtime.h>
#include <hip/hip_bf16.h>

typedef __bf16  v16bf __attribute__((ext_vector_type(16)));
typedef __bf16  v8bf  __attribute__((ext_vector_type(8)));
typedef float   v8f   __attribute__((ext_vector_type(8)));

#define BATCH 64
#define MAXLEN 64
#define VOCAB 40
#define CEMB 32

// ---------------------------------------------------------------------------
// fp32 -> bf16 conversion (one-time; weights then live in L2 at half size)
// ---------------------------------------------------------------------------
__global__ void conv_f32_bf16(const float* __restrict__ s, __bf16* __restrict__ d, int n) {
    int i = blockIdx.x * blockDim.x + threadIdx.x;
    if (i < n) d[i] = (__bf16)s[i];
}

// ---------------------------------------------------------------------------
// Initial hidden states: h = z @ fc_init_w^T + b, split into h0/h1/h2 (bf16)
// ---------------------------------------------------------------------------
__global__ void init_states(const float* __restrict__ z, const float* __restrict__ w,
                            const float* __restrict__ b,
                            __bf16* __restrict__ h0, __bf16* __restrict__ h1,
                            __bf16* __restrict__ h2) {
    int i = blockIdx.x * blockDim.x + threadIdx.x;
    if (i >= BATCH * 3584) return;
    int bi = i / 3584, j = i % 3584;
    const float* zr = z + (size_t)bi * 512;
    const float* wr = w + (size_t)j * 512;
    float s = b[j];
    for (int k = 0; k < 512; ++k) s += zr[k] * wr[k];
    if (j < 512)       h0[bi * 512  + j]          = (__bf16)s;
    else if (j < 1536) h1[bi * 1024 + (j - 512)]  = (__bf16)s;
    else               h2[bi * 2048 + (j - 1536)] = (__bf16)s;
}

__global__ void dec_init(int* __restrict__ dec, const int* __restrict__ st) {
    if (threadIdx.x < BATCH) dec[threadIdx.x] = st[0];
}

// ---------------------------------------------------------------------------
// Embedding gather for current tokens -> x0 [64,32] bf16
// ---------------------------------------------------------------------------
__global__ void embed_gather(const int* __restrict__ dec, const __bf16* __restrict__ embb,
                             __bf16* __restrict__ x0) {
    int i = blockIdx.x * blockDim.x + threadIdx.x;
    if (i >= BATCH * CEMB) return;
    int bi = i >> 5, k = i & 31;
    x0[i] = embb[dec[bi] * CEMB + k];
}

// ---------------------------------------------------------------------------
// GRU projection GEMMs via bf16 WMMA, fp32 accumulate.
//   blockIdx.z == 0 : GI = X @ Wih^T + bih    (K = In)
//   blockIdx.z == 1 : GH = H @ Whh^T + bhh    (K = H)
// One wave per 16-wide N column strip, covering ALL FOUR 16-row M tiles
// (whole batch of 64): the B (weight) fragment is loaded once and feeds
// 4 WMMAs, so per-step weight traffic from L2 is exactly one pass over
// the weights (49 MB bf16 total per timestep across the three layers).
//
// Fragment layouts used here (16-bit, wave32; contiguous K, same order for A and B):
//   A: lane = row (l%16); 16 contiguous K values starting at half*16.
//   B: lane holds B column n = nb + l  (== row n of W, K-contiguous),
//      16 contiguous K values starting at half*16.
//   C/D: VGPR v -> row M = v + half*8, column N = l.
// ---------------------------------------------------------------------------
__global__ __launch_bounds__(32)
void gru_gemm(const __bf16* __restrict__ X, const __bf16* __restrict__ Hst,
              const __bf16* __restrict__ Wih, const __bf16* __restrict__ Whh,
              const float* __restrict__ bih, const float* __restrict__ bhh,
              float* __restrict__ GI, float* __restrict__ GH, int In, int H) {
    const int nb   = blockIdx.x * 16;
    const bool rec = (blockIdx.z != 0);
    const int  N3  = 3 * H;
    const int  K   = rec ? H : In;

    const __bf16* A    = rec ? Hst : X;
    const __bf16* W    = rec ? Whh : Wih;
    const float*  bias = rec ? bhh : bih;
    float*        Out  = rec ? GH  : GI;

    const int lane = threadIdx.x & 31;
    const int half = lane >> 4;       // 0 for lanes 0-15, 1 for 16-31
    const int l    = lane & 15;

    const __bf16* abase = A + (size_t)l * K + half * 16;         // row l of M-tile 0
    const __bf16* brow  = W + (size_t)(nb + l) * K + half * 16;  // weight row = B column

    // Fold bias (depends only on column N = nb + l) into the accumulators.
    const float bi = bias[nb + l];
    v8f c[4];
#pragma unroll
    for (int m = 0; m < 4; ++m)
#pragma unroll
        for (int v = 0; v < 8; ++v) c[m][v] = bi;

    const size_t mstride = (size_t)16 * K;   // 16 rows per M tile
    for (int kb = 0; kb < K; kb += 32) {
        v16bf b = *(const v16bf*)(brow + kb);                    // 32B, one B fragment
#pragma unroll
        for (int m = 0; m < 4; ++m) {
            const __bf16* ar = abase + m * mstride + kb;
            v8bf a0 = *(const v8bf*)(ar);                        // 16B aligned
            v8bf a1 = *(const v8bf*)(ar + 8);                    // 16B aligned
            v16bf a = __builtin_shufflevector(a0, a1,
                        0, 1, 2, 3, 4, 5, 6, 7, 8, 9, 10, 11, 12, 13, 14, 15);
            c[m] = __builtin_amdgcn_wmma_f32_16x16x32_bf16(
                       /*neg_a=*/false, a, /*neg_b=*/false, b,
                       /*c_mod=*/(short)0, c[m], /*reuse_a=*/false, /*reuse_b=*/false);
        }
    }

#pragma unroll
    for (int m = 0; m < 4; ++m) {
        float* o = Out + (size_t)(m * 16 + half * 8) * N3 + nb + l;
#pragma unroll
        for (int v = 0; v < 8; ++v) o[(size_t)v * N3] = c[m][v];
    }
}

// ---------------------------------------------------------------------------
// GRU gate nonlinearity; updates hidden state in place (bf16).
// ---------------------------------------------------------------------------
__global__ void gru_gate(const float* __restrict__ GI, const float* __restrict__ GH,
                         __bf16* __restrict__ h, int H) {
    int i = blockIdx.x * blockDim.x + threadIdx.x;
    if (i >= BATCH * H) return;
    int bi = i / H, j = i % H;
    const float* gi = GI + (size_t)bi * 3 * H;
    const float* gh = GH + (size_t)bi * 3 * H;
    float r  = 1.f / (1.f + __expf(-(gi[j]         + gh[j])));
    float zg = 1.f / (1.f + __expf(-(gi[H + j]     + gh[H + j])));
    float n  = tanhf(gi[2 * H + j] + r * gh[2 * H + j]);
    float hp = (float)h[i];
    h[i] = (__bf16)((1.f - zg) * n + zg * hp);
}

// ---------------------------------------------------------------------------
// logits = h2 @ fc_out_w^T ; argmax -> token, feed back + record output.
// One block per batch element.
// ---------------------------------------------------------------------------
__global__ void out_argmax(const __bf16* __restrict__ h2, const __bf16* __restrict__ fo,
                           float* __restrict__ out, int* __restrict__ dec, int t) {
    __shared__ float lg[VOCAB];
    const int b = blockIdx.x, tid = threadIdx.x;
    if (tid < VOCAB) {
        const __bf16* hr = h2 + (size_t)b * 2048;
        const __bf16* wr = fo + (size_t)tid * 2048;
        float s = 0.f;
        for (int k = 0; k < 2048; ++k) s += (float)hr[k] * (float)wr[k];
        lg[tid] = s;
    }
    __syncthreads();
    if (tid == 0) {
        int best = 0; float bv = lg[0];
        for (int v = 1; v < VOCAB; ++v) if (lg[v] > bv) { bv = lg[v]; best = v; }
        out[(size_t)b * MAXLEN + t] = (float)best;  // toks layout [B, max_len, 1]
        dec[b] = best;
    }
}

// ---------------------------------------------------------------------------
extern "C" void kernel_launch(void* const* d_in, const int* in_sizes, int n_in,
                              void* d_out, int out_size, void* d_ws, size_t ws_size,
                              hipStream_t stream) {
    (void)in_sizes; (void)n_in; (void)out_size; (void)ws_size;

    const float* z   = (const float*)d_in[0];
    const float* emb = (const float*)d_in[1];
    const float* fiw = (const float*)d_in[2];
    const float* fib = (const float*)d_in[3];
    const float* fow = (const float*)d_in[4];
    const float* wih[3] = {(const float*)d_in[5],  (const float*)d_in[9],  (const float*)d_in[13]};
    const float* whh[3] = {(const float*)d_in[6],  (const float*)d_in[10], (const float*)d_in[14]};
    const float* bih[3] = {(const float*)d_in[7],  (const float*)d_in[11], (const float*)d_in[15]};
    const float* bhh[3] = {(const float*)d_in[8],  (const float*)d_in[12], (const float*)d_in[16]};
    const int* startTok = (const int*)d_in[18];
    float* out = (float*)d_out;

    const int Hd[3] = {512, 1024, 2048};
    const int In[3] = {32, 512, 1024};

    char* ws = (char*)d_ws;
    size_t off = 0;
    auto alloc = [&](size_t bytes) -> char* {
        char* p = ws + off;
        off += (bytes + 255) & ~(size_t)255;
        return p;
    };

    __bf16* wihb[3]; __bf16* whhb[3]; __bf16* h[3];
    for (int li = 0; li < 3; ++li) {
        wihb[li] = (__bf16*)alloc((size_t)3 * Hd[li] * In[li] * 2);
        whhb[li] = (__bf16*)alloc((size_t)3 * Hd[li] * Hd[li] * 2);
    }
    __bf16* fob  = (__bf16*)alloc((size_t)VOCAB * 2048 * 2);
    __bf16* embb = (__bf16*)alloc((size_t)VOCAB * CEMB * 2);
    for (int li = 0; li < 3; ++li) h[li] = (__bf16*)alloc((size_t)BATCH * Hd[li] * 2);
    __bf16* x0 = (__bf16*)alloc((size_t)BATCH * CEMB * 2);
    float* GI  = (float*)alloc((size_t)BATCH * 6144 * 4);
    float* GH  = (float*)alloc((size_t)BATCH * 6144 * 4);
    int* dec   = (int*)alloc(BATCH * 4);

    // --- one-time setup: bf16 weight images + initial states ---
    for (int li = 0; li < 3; ++li) {
        int ni = 3 * Hd[li] * In[li];
        int nh = 3 * Hd[li] * Hd[li];
        conv_f32_bf16<<<(ni + 255) / 256, 256, 0, stream>>>(wih[li], wihb[li], ni);
        conv_f32_bf16<<<(nh + 255) / 256, 256, 0, stream>>>(whh[li], whhb[li], nh);
    }
    conv_f32_bf16<<<(VOCAB * 2048 + 255) / 256, 256, 0, stream>>>(fow, fob, VOCAB * 2048);
    conv_f32_bf16<<<(VOCAB * CEMB + 255) / 256, 256, 0, stream>>>(emb, embb, VOCAB * CEMB);
    init_states<<<(BATCH * 3584 + 255) / 256, 256, 0, stream>>>(z, fiw, fib, h[0], h[1], h[2]);
    dec_init<<<1, 64, 0, stream>>>(dec, startTok);

    // --- autoregressive decode: 64 sequential steps ---
    for (int t = 0; t < MAXLEN; ++t) {
        embed_gather<<<(BATCH * CEMB + 255) / 256, 256, 0, stream>>>(dec, embb, x0);
        for (int li = 0; li < 3; ++li) {
            const __bf16* X = (li == 0) ? x0 : h[li - 1];
            dim3 grid(3 * Hd[li] / 16, 1, 2);   // one wave covers all 4 M tiles
            gru_gemm<<<grid, 32, 0, stream>>>(X, h[li], wihb[li], whhb[li],
                                              bih[li], bhh[li], GI, GH, In[li], Hd[li]);
            gru_gate<<<(BATCH * Hd[li] + 255) / 256, 256, 0, stream>>>(GI, GH, h[li], Hd[li]);
        }
        out_argmax<<<BATCH, 64, 0, stream>>>(h[2], fob, out, dec, t);
    }
}
